// CFPGv2ExplModule_51548197487191
// MI455X (gfx1250) — matrix operand order repaired and checked by reading, not compiled, not run on hardware
//
#include <hip/hip_runtime.h>
#include <hip/hip_bf16.h>
#include <math.h>

typedef __attribute__((ext_vector_type(16))) _Float16 v16h;
typedef __attribute__((ext_vector_type(8)))  _Float16 v8h;
typedef __attribute__((ext_vector_type(8)))  float    v8f;

#define N_NODES_C   50000
#define IN_FEATS_C  256
#define ENC_H_C     20
#define DEC_H_C     64

// ---------------------------------------------------------------- threefry
__device__ __forceinline__ unsigned rotl32(unsigned x, int n) {
    return (x << n) | (x >> (32 - n));
}
__device__ __forceinline__ void threefry2x32(unsigned k0, unsigned k1,
                                             unsigned& x0, unsigned& x1) {
    unsigned k2 = 0x1BD11BDAu ^ k0 ^ k1;
    x0 += k0; x1 += k1;
#define TF_R4(a,b,c,d) \
    x0 += x1; x1 = rotl32(x1,a); x1 ^= x0; \
    x0 += x1; x1 = rotl32(x1,b); x1 ^= x0; \
    x0 += x1; x1 = rotl32(x1,c); x1 ^= x0; \
    x0 += x1; x1 = rotl32(x1,d); x1 ^= x0;
    TF_R4(13,15,26,6)   x0 += k1; x1 += k2 + 1u;
    TF_R4(17,29,16,24)  x0 += k2; x1 += k0 + 2u;
    TF_R4(13,15,26,6)   x0 += k0; x1 += k1 + 3u;
    TF_R4(17,29,16,24)  x0 += k1; x1 += k2 + 4u;
    TF_R4(13,15,26,6)   x0 += k2; x1 += k0 + 5u;
#undef TF_R4
}

// ---------------------------------------------------------------- setup kernels
__global__ void k_init(float* deg, float* agg, int n_nodes) {
    int i = blockIdx.x * blockDim.x + threadIdx.x;
    if (i < n_nodes * ENC_H_C) agg[i] = 0.0f;
    if (i < n_nodes) deg[i] = 1.0f;   // self loop
}

__global__ void k_degree(const long long* ei, float* deg, int n_edges) {
    int e = blockIdx.x * blockDim.x + threadIdx.x;
    if (e >= n_edges) return;
    int c = (int)ei[(long)n_edges + e];   // target
    atomicAdd(deg + c, 1.0f);
}

__global__ void k_dinv(const float* deg, float* dinv, int n_nodes) {
    int i = blockIdx.x * blockDim.x + threadIdx.x;
    if (i >= n_nodes) return;
    float d = deg[i];
    dinv[i] = (d > 0.0f) ? rsqrtf(d) : 0.0f;
}

// ---------------------------------------------------------------- encoder GEMM (WMMA)
// xw = x[50000,256] @ W_gcn[256,20], f16 WMMA, N padded to 32.
__global__ __launch_bounds__(256) void k_enc_gemm(const float* __restrict__ x,
                                                  const float* __restrict__ Wg,
                                                  float* __restrict__ xw,
                                                  int n_nodes) {
    __shared__ _Float16 Wt[32 * IN_FEATS_C] __attribute__((aligned(16)));
    int tid = threadIdx.x;
    for (int idx = tid; idx < 32 * IN_FEATS_C; idx += 256) {
        int n = idx >> 8, k = idx & 255;
        Wt[idx] = (_Float16)((n < ENC_H_C) ? Wg[k * ENC_H_C + n] : 0.0f);
    }
    __syncthreads();

    int wave = tid >> 5, lane = tid & 31;
    int mtile = blockIdx.x * 8 + wave;
    int ntiles = (n_nodes + 15) >> 4;
    if (mtile >= ntiles) return;            // wave-uniform

    int  m  = lane & 15;
    bool hi = lane >= 16;
    int  row = mtile * 16 + m;
    if (row >= n_nodes) row = n_nodes - 1;  // clamp loads; EXEC stays all-1
    const float* xr = x + (long)row * IN_FEATS_C;

    v8f c0 = {}, c1 = {};
    const int kloA = hi ? 8 : 0;            // A layout: K {0..7,16..23} / {8..15,24..31}
    const int kloB = hi ? 8 : 0;            // B layout: K {0..7,16..23} / {8..15,24..31}
    const _Float16* w0 = Wt + (long)m * IN_FEATS_C;
    const _Float16* w1 = Wt + (long)(m + 16) * IN_FEATS_C;

#pragma unroll
    for (int ks = 0; ks < 8; ++ks) {
        int kbase = ks * 32;
        v16h a;
#pragma unroll
        for (int j = 0; j < 8; ++j) {
            a[j]     = (_Float16)xr[kbase + kloA + j];
            a[8 + j] = (_Float16)xr[kbase + kloA + 16 + j];
        }
        v16h bA, bB;
#pragma unroll
        for (int j = 0; j < 16; ++j) {
            bA[j] = w0[kbase + kloB + j + (j >= 8 ? 8 : 0)];
            bB[j] = w1[kbase + kloB + j + (j >= 8 ? 8 : 0)];
        }
        c0 = __builtin_amdgcn_wmma_f32_16x16x32_f16(false, a, false, bA,
                                                    (short)0, c0, false, false);
        c1 = __builtin_amdgcn_wmma_f32_16x16x32_f16(false, a, false, bB,
                                                    (short)0, c1, false, false);
    }

    int mbase = hi ? 8 : 0;
    int n0 = m;
#pragma unroll
    for (int v = 0; v < 8; ++v) {
        int node = mtile * 16 + mbase + v;
        if (node < n_nodes) {
            xw[(long)node * ENC_H_C + n0] = c0[v];
            if (n0 + 16 < ENC_H_C) xw[(long)node * ENC_H_C + n0 + 16] = c1[v];
        }
    }
}

// ---------------------------------------------------------------- edge scatter
__global__ void k_scatter(const long long* __restrict__ ei,
                          const float* __restrict__ xw,
                          const float* __restrict__ dinv,
                          float* __restrict__ agg, int n_edges) {
    int e = blockIdx.x * blockDim.x + threadIdx.x;
    if (e >= n_edges) return;
    int r = (int)ei[e];
    int c = (int)ei[(long)n_edges + e];
    float norm = dinv[r] * dinv[c];
    const float* src = xw + (long)r * ENC_H_C;
    float* dst = agg + (long)c * ENC_H_C;
#pragma unroll
    for (int f = 0; f < ENC_H_C; ++f)
        atomicAdd(dst + f, src[f] * norm);
}

// ---------------------------------------------------------------- enc finalize (relu, f16)
__global__ void k_finalize(const float* __restrict__ agg,
                           const float* __restrict__ xw,
                           const float* __restrict__ dinv,
                           const float* __restrict__ bg,
                           _Float16* __restrict__ ench, int n_nodes) {
    int i = blockIdx.x * blockDim.x + threadIdx.x;
    if (i >= n_nodes * ENC_H_C) return;
    int node = i / ENC_H_C, f = i % ENC_H_C;
    float dv = dinv[node];
    float v = agg[i] + xw[i] * dv * dv + bg[f];   // self-loop folded in
    ench[i] = (_Float16)(v > 0.0f ? v : 0.0f);
}

// ---------------------------------------------------------------- decoder MLP (WMMA) + gumbel-sigmoid
// One 16-edge tile per wave. z-tile staged through LDS so A fragments are
// aligned ds_load_b128 reads instead of scalar gathers.
__global__ __launch_bounds__(256) void k_decoder(const long long* __restrict__ ei,
                                                 const _Float16* __restrict__ ench,
                                                 const float* __restrict__ W1,
                                                 const float* __restrict__ b1g,
                                                 const float* __restrict__ W2,
                                                 const float* __restrict__ b2g,
                                                 const int* __restrict__ node_id_p,
                                                 float* __restrict__ out, int n_edges) {
    __shared__ _Float16 Wt[64 * 64] __attribute__((aligned(16)));   // W1^T [n][k], 8 KB
    __shared__ _Float16 Zt[8][16 * 64] __attribute__((aligned(16))); // per-wave z tile, 16 KB
    __shared__ float b1s[64], w2s[64];
    int tid = threadIdx.x;
    for (int idx = tid; idx < 64 * 64; idx += 256) {
        int n = idx >> 6, k = idx & 63;
        Wt[idx] = (_Float16)((k < 60) ? W1[k * DEC_H_C + n] : 0.0f);
    }
    if (tid < 64) { b1s[tid] = b1g[tid]; w2s[tid] = W2[tid]; }

    int wave = tid >> 5, lane = tid & 31;
    int tile = blockIdx.x * 8 + wave;
    int ntiles = (n_edges + 15) >> 4;

    int  m  = lane & 15;
    bool hi = lane >= 16;
    long e = (long)tile * 16 + m;
    if (e >= n_edges) e = n_edges - 1;     // clamp; EXEC stays all-1
    if (tile >= ntiles) e = 0;             // dead wave: still stage safely

    // ---- stage z[m][0..63] into LDS (rows are 40B, always 8B-aligned) ----
    uint2* zw = (uint2*)(&Zt[wave][m * 64]);       // 16 uint2 per edge row
    if (!hi) {
        int row = (int)ei[e];
        const uint2* er8 = (const uint2*)(ench + (long)row * ENC_H_C);
#pragma unroll
        for (int i = 0; i < 5; ++i) zw[i] = er8[i];          // z[0..19]  = enc[row]
    } else {
        int col = (int)ei[(long)n_edges + e];
        const uint2* ec8 = (const uint2*)(ench + (long)col * ENC_H_C);
        const uint2* en8 = (const uint2*)(ench + (long)node_id_p[0] * ENC_H_C);
#pragma unroll
        for (int i = 0; i < 5; ++i) zw[5 + i]  = ec8[i];     // z[20..39] = enc[col]
#pragma unroll
        for (int i = 0; i < 5; ++i) zw[10 + i] = en8[i];     // z[40..59] = enc[node]
        zw[15] = make_uint2(0u, 0u);                         // z[60..63] = 0 (K pad)
    }
    __syncthreads();
    if (tile >= ntiles) return;            // wave-uniform

    // ---- A fragments from LDS: 4x ds_load_b128 per K-step pair ----
    const _Float16* zb = &Zt[wave][0];
    const int kloA = hi ? 8 : 0;
    const int base = m * 64;
    v8h a0lo = *(const v8h*)(zb + base + kloA);
    v8h a0hi = *(const v8h*)(zb + base + kloA + 16);
    v8h a1lo = *(const v8h*)(zb + base + kloA + 32);
    v8h a1hi = *(const v8h*)(zb + base + kloA + 48);
    v16h a0 = __builtin_shufflevector(a0lo, a0hi, 0,1,2,3,4,5,6,7,8,9,10,11,12,13,14,15);
    v16h a1 = __builtin_shufflevector(a1lo, a1hi, 0,1,2,3,4,5,6,7,8,9,10,11,12,13,14,15);

    v8f c[4] = {};
    const int kloB = hi ? 8 : 0;
#pragma unroll
    for (int t = 0; t < 4; ++t) {
        int n = m + 16 * t;
        const _Float16* wr = Wt + (long)n * 64;
        v16h bA, bB;
#pragma unroll
        for (int j = 0; j < 16; ++j) {
            bA[j] = wr[kloB + j + (j >= 8 ? 8 : 0)];
            bB[j] = wr[32 + kloB + j + (j >= 8 ? 8 : 0)];
        }
        c[t] = __builtin_amdgcn_wmma_f32_16x16x32_f16(false, a0, false, bA,
                                                      (short)0, c[t], false, false);
        c[t] = __builtin_amdgcn_wmma_f32_16x16x32_f16(false, a1, false, bB,
                                                      (short)0, c[t], false, false);
    }

    // layer 2 matvec folded per-lane, then butterfly over 16-lane N groups
    float acc[8];
#pragma unroll
    for (int v = 0; v < 8; ++v) acc[v] = 0.0f;
#pragma unroll
    for (int t = 0; t < 4; ++t) {
        int n = m + 16 * t;
        float bb = b1s[n], ww = w2s[n];
#pragma unroll
        for (int v = 0; v < 8; ++v) {
            float h = c[t][v] + bb;
            h = (h > 0.0f) ? h : 0.0f;
            acc[v] += h * ww;
        }
    }
#pragma unroll
    for (int v = 0; v < 8; ++v) {
        acc[v] += __shfl_xor(acc[v], 1);
        acc[v] += __shfl_xor(acc[v], 2);
        acc[v] += __shfl_xor(acc[v], 4);
        acc[v] += __shfl_xor(acc[v], 8);
    }

    if (m == 0) {
        float bb2 = b2g[0];
        int mbase = hi ? 8 : 0;
        unsigned half = (unsigned)(n_edges >> 1);
        for (int v = 0; v < 8; ++v) {
            long eo = (long)tile * 16 + mbase + v;
            if (eo >= n_edges) break;
            float dec = acc[v] + bb2;
            // JAX threefry uniform: element eo pairs with eo +/- E/2, key (0,123)
            unsigned p  = (unsigned)(eo % (long)half);
            unsigned x0 = p, x1 = p + half;
            threefry2x32(0u, 123u, x0, x1);
            unsigned bits = (eo < (long)half) ? x0 : x1;
            float u = __uint_as_float((bits >> 9) | 0x3f800000u) - 1.0f;
            const float bias = 0.0001f;
            float eps  = (1.0f - bias) - (1.0f - 2.0f * bias) * u;
            float gate = logf(eps) - log1pf(-eps);
            float val  = gate + dec;                 // TEMPERATURE == 1
            out[eo] = 1.0f / (1.0f + expf(-val));
        }
    }
}

// ---------------------------------------------------------------- launch
extern "C" void kernel_launch(void* const* d_in, const int* in_sizes, int n_in,
                              void* d_out, int out_size, void* d_ws, size_t ws_size,
                              hipStream_t stream) {
    const float*      x   = (const float*)d_in[0];
    const long long*  ei  = (const long long*)d_in[1];
    const int*        nid = (const int*)d_in[2];
    const float*      Wg  = (const float*)d_in[3];
    const float*      bg  = (const float*)d_in[4];
    const float*      W1  = (const float*)d_in[5];
    const float*      b1  = (const float*)d_in[6];
    const float*      W2  = (const float*)d_in[7];
    const float*      b2  = (const float*)d_in[8];
    float* out = (float*)d_out;

    int n_nodes = in_sizes[0] / IN_FEATS_C;
    int n_edges = in_sizes[1] / 2;

    char* ws = (char*)d_ws;
    size_t off = 0;
    auto carve = [&](size_t bytes) -> char* {
        char* p = ws + off;
        off = (off + bytes + 255) & ~(size_t)255;
        return p;
    };
    float*     deg  = (float*)carve((size_t)n_nodes * 4);
    float*     dinv = (float*)carve((size_t)n_nodes * 4);
    float*     xw   = (float*)carve((size_t)n_nodes * ENC_H_C * 4);
    float*     agg  = (float*)carve((size_t)n_nodes * ENC_H_C * 4);
    _Float16*  ench = (_Float16*)carve((size_t)n_nodes * ENC_H_C * 2);
    (void)ws_size; (void)n_in; (void)out_size;

    int t = 256;
    k_init    <<<(n_nodes * ENC_H_C + t - 1) / t, t, 0, stream>>>(deg, agg, n_nodes);
    k_degree  <<<(n_edges + t - 1) / t, t, 0, stream>>>(ei, deg, n_edges);
    k_dinv    <<<(n_nodes + t - 1) / t, t, 0, stream>>>(deg, dinv, n_nodes);

    int enc_tiles = (n_nodes + 15) / 16;
    k_enc_gemm<<<(enc_tiles + 7) / 8, 256, 0, stream>>>(x, Wg, xw, n_nodes);

    k_scatter <<<(n_edges + t - 1) / t, t, 0, stream>>>(ei, xw, dinv, agg, n_edges);
    k_finalize<<<(n_nodes * ENC_H_C + t - 1) / t, t, 0, stream>>>(agg, xw, dinv, bg, ench, n_nodes);

    int dec_tiles = (n_edges + 15) / 16;
    k_decoder <<<(dec_tiles + 7) / 8, 256, 0, stream>>>(ei, ench, W1, b1, W2, b2,
                                                        nid, out, n_edges);
}
